// MultiHeadAttention_86612310491639
// MI455X (gfx1250) — matrix operand order repaired and checked by reading, not compiled, not run on hardware
//
#include <hip/hip_runtime.h>
#include <hip/hip_bf16.h>

typedef __attribute__((ext_vector_type(16))) __bf16 v16bf;
typedef __attribute__((ext_vector_type(8)))  __bf16 v8bf;
typedef __attribute__((ext_vector_type(8)))  float  v8f;
typedef __attribute__((ext_vector_type(4)))  unsigned int u32x4;
typedef __attribute__((ext_vector_type(8)))  int i32x8;
typedef __attribute__((ext_vector_type(4)))  int i32x4;

#define S_LEN   4096
#define DMODEL  512
#define NHEAD   8
#define HDIM    64
#define BATCH   2
#define LOG2E   1.4426950408889634f
#define SCALE_F 0.04419417382415922f     /* 1/sqrt(512) */
#define NEG2_F  (-1.4426950408889634e9f) /* -1e9 * log2e : mask bias in exp2 domain */

/* ---------------- conversion kernels ---------------- */

__global__ void convert_f32_bf16(const float* __restrict__ in, __bf16* __restrict__ out, int n) {
    int i = blockIdx.x * blockDim.x + threadIdx.x;
    if (i < n) out[i] = (__bf16)in[i];
}

__global__ void mask_to_bias(const int* __restrict__ mask, float* __restrict__ bias, int n) {
    int i = blockIdx.x * blockDim.x + threadIdx.x;
    if (i < n) bias[i] = mask[i] ? NEG2_F : 0.0f;
}

/* W[K][N] row-major (fp32)  ->  Wt[N][K] (bf16) */
__global__ void transpose_to_bf16(const float* __restrict__ W, __bf16* __restrict__ Wt, int K, int N) {
    __shared__ float tile[32][33];
    int k0 = blockIdx.y * 32, n0 = blockIdx.x * 32;
    int tx = threadIdx.x & 31, ty = threadIdx.x >> 5;   // 32 x 8
    #pragma unroll
    for (int r = ty; r < 32; r += 8) tile[r][tx] = W[(size_t)(k0 + r) * N + n0 + tx];
    __syncthreads();
    #pragma unroll
    for (int r = ty; r < 32; r += 8) Wt[(size_t)(n0 + r) * K + k0 + tx] = (__bf16)tile[tx][r];
}

/* ---------------- bf16 WMMA GEMM: C = (A[M,K] * Bt[N,K]^T + bias) * outScale ----
 * wave tile 32x64 (2x4 accumulators), workgroup tile 128x128 (8 waves, 4m x 2n)
 * MODE 0: bf16 row-major; MODE 1: bf16 per-head transposed (V); MODE 2: fp32
 */
template<int MODE>
__global__ __launch_bounds__(256)
void gemm_bf16_wmma(const __bf16* __restrict__ A, const __bf16* __restrict__ Bt,
                    const float* __restrict__ bias, void* __restrict__ Cout,
                    float outScale, int M, int N, int K)
{
    const int lane = threadIdx.x & 31;
    const int wave = threadIdx.x >> 5;
    const int lid  = lane & 15;
    const int hi16 = lane >> 4;
    const int wm   = wave >> 1;
    const int wn   = wave & 1;
    const int m_base = blockIdx.y * 128 + wm * 32;
    const int n_base = blockIdx.x * 128 + wn * 64;

    v8f acc[2][4] = {};
    union AF { v16bf v; struct { v8bf lo, hi; } h; };

    for (int k0 = 0; k0 < K; k0 += 32) {
        AF a[2];
        #pragma unroll
        for (int i = 0; i < 2; ++i) {
            const __bf16* p = A + (size_t)(m_base + i*16 + lid) * K + k0 + hi16*16;
            a[i].h.lo = *(const v8bf*)p;
            a[i].h.hi = *(const v8bf*)(p + 8);
        }
        v16bf bfr[4];
        #pragma unroll
        for (int j = 0; j < 4; ++j)
            bfr[j] = *(const v16bf*)(Bt + (size_t)(n_base + j*16 + lid) * K + k0 + hi16*16);
        #pragma unroll
        for (int i = 0; i < 2; ++i)
            #pragma unroll
            for (int j = 0; j < 4; ++j)
                acc[i][j] = __builtin_amdgcn_wmma_f32_16x16x32_bf16(
                    false, a[i].v, false, bfr[j], (short)0, acc[i][j], false, false);
    }

    #pragma unroll
    for (int i = 0; i < 2; ++i)
        #pragma unroll
        for (int j = 0; j < 4; ++j)
            #pragma unroll
            for (int e = 0; e < 8; ++e) {
                int row = m_base + i*16 + e + hi16*8;
                int col = n_base + j*16 + lid;
                float v = (acc[i][j][e] + bias[col]) * outScale;
                if (MODE == 0) {
                    ((__bf16*)Cout)[(size_t)row * N + col] = (__bf16)v;
                } else if (MODE == 1) {
                    int gb = row >> 12, gs = row & (S_LEN - 1);
                    int gh = col >> 6,  gd = col & (HDIM - 1);
                    ((__bf16*)Cout)[(((size_t)(gb * NHEAD + gh) * HDIM + gd) << 12) + gs] = (__bf16)v;
                } else {
                    ((float*)Cout)[(size_t)row * N + col] = v;
                }
            }
}

/* ---- TDM: async 2-D tile load (64 rows x 64 bf16) from global into LDS ---- */
__device__ __forceinline__ void tdm_load_64x64_bf16(unsigned lds_addr, const void* gptr,
                                                    unsigned row_stride_elems)
{
    unsigned long long ga = (unsigned long long)(uintptr_t)gptr;
    u32x4 g0;
    g0[0] = 1u;                                              // count=1, user descriptor
    g0[1] = lds_addr;                                        // LDS byte address
    g0[2] = (unsigned)ga;                                    // global addr [31:0]
    g0[3] = ((unsigned)(ga >> 32) & 0x01FFFFFFu) | 0x80000000u; // addr[56:32] | type=2
    i32x8 g1;
    g1[0] = (int)(1u << 16);        // workgroup_mask=0, data_size=1 (2 bytes), flags=0
    g1[1] = (int)(64u << 16);       // tensor_dim0[15:0]=64 (atomic_barrier_addr=0)
    g1[2] = (int)(64u << 16);       // tensor_dim0[31:16]=0, tensor_dim1[15:0]=64
    g1[3] = (int)(64u << 16);       // tensor_dim1[31:16]=0, tile_dim0=64
    g1[4] = 64;                     // tile_dim1=64, tile_dim2=0 (2-D)
    g1[5] = (int)row_stride_elems;  // tensor_dim0_stride[31:0] (element units)
    g1[6] = 0;                      // d0_stride[47:32]=0, d1_stride[15:0]=0
    g1[7] = 0;                      // d1_stride[47:16]=0
    i32x4 gz4 = {0, 0, 0, 0};       // D# groups 2/3 unused for 2-D tiles
    i32x8 gz8 = {0, 0, 0, 0, 0, 0, 0, 0};
    __builtin_amdgcn_tensor_load_to_lds(g0, g1, gz4, gz4, gz8, 0);
}

/* ---------------- flash attention (one wave = 16 queries, 64-key chunks) ------
 * Qb (pre-scaled by SCALE*log2e), Kb: bf16 [B, S, DMODEL]
 * Vt: bf16 [B, H, HDIM, S];  biasM: float [B, S] (0 or -1e9*log2e)
 * K/V chunks staged into LDS by the Tensor Data Mover, double-buffered.
 */
__global__ __launch_bounds__(256)
void flash_attn_wmma(const __bf16* __restrict__ Qb, const __bf16* __restrict__ Kb,
                     const __bf16* __restrict__ Vt, const float* __restrict__ biasM,
                     __bf16* __restrict__ Ob)
{
    __shared__ __align__(128) __bf16 kbuf[2][64][64];     // [key][dim]   2 x 8 KB
    __shared__ __align__(128) __bf16 vbuf[2][64][64];     // [dim][key]   2 x 8 KB
    __shared__ __align__(128) __bf16 pstage[8][16][64];   // per-wave P   16 KB

    const int lane = threadIdx.x & 31;
    const int wave = threadIdx.x >> 5;
    const int lid  = lane & 15;
    const int hi16 = lane >> 4;
    const int bh = blockIdx.y;
    const int b  = bh >> 3;
    const int h  = bh & 7;
    const int q0 = blockIdx.x * 128 + wave * 16;

    union AF { v16bf v; struct { v8bf lo, hi; } h; };

    AF aQ[2];
    {
        const __bf16* qp = Qb + (size_t)(b * S_LEN + q0 + lid) * DMODEL + h * HDIM;
        #pragma unroll
        for (int t = 0; t < 2; ++t) {
            const __bf16* p = qp + t*32 + hi16*16;
            aQ[t].h.lo = *(const v8bf*)p;
            aQ[t].h.hi = *(const v8bf*)(p + 8);
        }
    }

    const __bf16* kgbase = Kb + (size_t)b * S_LEN * DMODEL + h * HDIM;   // + key*DMODEL
    const __bf16* vgbase = Vt + ((size_t)bh * HDIM << 12);               // + d*S_LEN + key
    unsigned koff[2], voff[2];
    #pragma unroll
    for (int p = 0; p < 2; ++p) {
        koff[p] = (unsigned)(uintptr_t)&kbuf[p][0][0];
        voff[p] = (unsigned)(uintptr_t)&vbuf[p][0][0];
    }

    if (wave == 0) {    // prefetch chunk 0 (tensor ops issue once per wave, EXEC ignored)
        tdm_load_64x64_bf16(koff[0], kgbase, DMODEL);
        tdm_load_64x64_bf16(voff[0], vgbase, S_LEN);
    }

    float m_run[8], l_run[8];   // l_run: per-lane partial sum (reduced once at end)
    #pragma unroll
    for (int e = 0; e < 8; ++e) { m_run[e] = -3.0e38f; l_run[e] = 0.0f; }
    v8f o[4] = {};

    const float* brow = biasM + b * S_LEN;

    for (int it = 0; it < S_LEN / 64; ++it) {
        const int kc  = it * 64;
        const int cur = it & 1;

        __syncthreads();                       // buf[cur^1] no longer being read
        if (wave == 0) {
            if (it + 1 < S_LEN / 64) {
                tdm_load_64x64_bf16(koff[cur ^ 1], kgbase + (size_t)(kc + 64) * DMODEL, DMODEL);
                tdm_load_64x64_bf16(voff[cur ^ 1], vgbase + (kc + 64), S_LEN);
                __builtin_amdgcn_s_wait_tensorcnt(2);  // in-order => chunk `it` resident
            } else {
                __builtin_amdgcn_s_wait_tensorcnt(0);
            }
        }
        __syncthreads();                       // chunk `it` visible to all waves

        float kb[4];
        #pragma unroll
        for (int t = 0; t < 4; ++t) kb[t] = brow[kc + t*16 + lid];

        v8f s[4];
        #pragma unroll
        for (int jn = 0; jn < 4; ++jn) {
            const __bf16* kp = &kbuf[cur][jn*16 + lid][hi16*16];
            v16bf b0 = *(const v16bf*)kp;
            v16bf b1 = *(const v16bf*)(kp + 32);
            v8f z = {};
            z     = __builtin_amdgcn_wmma_f32_16x16x32_bf16(false, aQ[0].v, false, b0, (short)0, z, false, false);
            s[jn] = __builtin_amdgcn_wmma_f32_16x16x32_bf16(false, aQ[1].v, false, b1, (short)0, z, false, false);
        }

        /* add mask bias per accumulator (consecutive VGPRs -> packed adds, no gathers) */
        #pragma unroll
        for (int jn = 0; jn < 4; ++jn)
            #pragma unroll
            for (int e = 0; e < 8; ++e) s[jn][e] += kb[jn];

        float alpha[8];
        #pragma unroll
        for (int e = 0; e < 8; ++e) {
            float cm = fmaxf(fmaxf(s[0][e], s[1][e]), fmaxf(s[2][e], s[3][e]));
            #pragma unroll
            for (int d = 1; d < 16; d <<= 1) cm = fmaxf(cm, __shfl_xor(cm, d, 32));
            float nm = fmaxf(m_run[e], cm);
            alpha[e] = __builtin_amdgcn_exp2f(m_run[e] - nm);
            m_run[e] = nm;
            float p0 = __builtin_amdgcn_exp2f(s[0][e] - nm);
            float p1 = __builtin_amdgcn_exp2f(s[1][e] - nm);
            float p2 = __builtin_amdgcn_exp2f(s[2][e] - nm);
            float p3 = __builtin_amdgcn_exp2f(s[3][e] - nm);
            l_run[e] = l_run[e] * alpha[e] + ((p0 + p1) + (p2 + p3));
            int row = e + hi16*8;
            pstage[wave][row][lid]      = (__bf16)p0;
            pstage[wave][row][16 + lid] = (__bf16)p1;
            pstage[wave][row][32 + lid] = (__bf16)p2;
            pstage[wave][row][48 + lid] = (__bf16)p3;
        }
        #pragma unroll
        for (int j = 0; j < 4; ++j)
            #pragma unroll
            for (int e = 0; e < 8; ++e) o[j][e] *= alpha[e];

        /* P (C-layout) staged in LDS -> A-fragment layout; DS ops in-order per wave */
        #pragma unroll
        for (int t = 0; t < 2; ++t) {
            AF aP;
            aP.h.lo = *(const v8bf*)&pstage[wave][lid][t*32 + hi16*16];
            aP.h.hi = *(const v8bf*)&pstage[wave][lid][t*32 + hi16*16 + 8];
            #pragma unroll
            for (int j = 0; j < 4; ++j) {
                v16bf bV = *(const v16bf*)&vbuf[cur][j*16 + lid][t*32 + hi16*16];
                o[j] = __builtin_amdgcn_wmma_f32_16x16x32_bf16(false, aP.v, false, bV, (short)0, o[j], false, false);
            }
        }
    }

    float inv[8];
    #pragma unroll
    for (int e = 0; e < 8; ++e) {
        float t = l_run[e];
        #pragma unroll
        for (int d = 1; d < 16; d <<= 1) t += __shfl_xor(t, d, 32);
        inv[e] = 1.0f / t;
    }
    #pragma unroll
    for (int j = 0; j < 4; ++j)
        #pragma unroll
        for (int e = 0; e < 8; ++e) {
            int row = q0 + e + hi16*8;
            int col = h * HDIM + j*16 + lid;
            Ob[(size_t)(b * S_LEN + row) * DMODEL + col] = (__bf16)(o[j][e] * inv[e]);
        }
}

/* ---------------- launcher ---------------- */

extern "C" void kernel_launch(void* const* d_in, const int* in_sizes, int n_in,
                              void* d_out, int out_size, void* d_ws, size_t ws_size,
                              hipStream_t stream)
{
    const float* x  = (const float*)d_in[0];
    const int* mask = (const int*)  d_in[1];
    const float* Wq = (const float*)d_in[2];
    const float* bq = (const float*)d_in[3];
    const float* Wk = (const float*)d_in[4];
    const float* bk = (const float*)d_in[5];
    const float* Wv = (const float*)d_in[6];
    const float* bv = (const float*)d_in[7];
    const float* Wo = (const float*)d_in[8];
    const float* bo = (const float*)d_in[9];

    const int M = BATCH * S_LEN;                                    // 8192
    const size_t xbytes = (size_t)M * DMODEL * sizeof(__bf16);      // 8 MB
    const size_t wbytes = (size_t)DMODEL * DMODEL * sizeof(__bf16); // 512 KB

    char* ws = (char*)d_ws;
    __bf16* Xb  = (__bf16*)ws; ws += xbytes;
    __bf16* Wqt = (__bf16*)ws; ws += wbytes;
    __bf16* Wkt = (__bf16*)ws; ws += wbytes;
    __bf16* Wvt = (__bf16*)ws; ws += wbytes;
    __bf16* Wot = (__bf16*)ws; ws += wbytes;
    __bf16* Qb  = (__bf16*)ws; ws += xbytes;
    __bf16* Kb  = (__bf16*)ws; ws += xbytes;
    __bf16* Vt  = (__bf16*)ws; ws += xbytes;
    __bf16* AO  = (__bf16*)ws; ws += xbytes;
    float*  biasM = (float*)ws; ws += (size_t)BATCH * S_LEN * sizeof(float);

    convert_f32_bf16<<<(M * DMODEL + 255) / 256, 256, 0, stream>>>(x, Xb, M * DMODEL);
    mask_to_bias<<<(BATCH * S_LEN + 255) / 256, 256, 0, stream>>>(mask, biasM, BATCH * S_LEN);

    dim3 tgrid(DMODEL / 32, DMODEL / 32);
    transpose_to_bf16<<<tgrid, 256, 0, stream>>>(Wq, Wqt, DMODEL, DMODEL);
    transpose_to_bf16<<<tgrid, 256, 0, stream>>>(Wk, Wkt, DMODEL, DMODEL);
    transpose_to_bf16<<<tgrid, 256, 0, stream>>>(Wv, Wvt, DMODEL, DMODEL);
    transpose_to_bf16<<<tgrid, 256, 0, stream>>>(Wo, Wot, DMODEL, DMODEL);

    const float qscale = SCALE_F * LOG2E;   // fold score scale + log2e into Q
    dim3 ggrid(DMODEL / 128, M / 128);
    gemm_bf16_wmma<0><<<ggrid, 256, 0, stream>>>(Xb, Wqt, bq, (void*)Qb, qscale, M, DMODEL, DMODEL);
    gemm_bf16_wmma<0><<<ggrid, 256, 0, stream>>>(Xb, Wkt, bk, (void*)Kb, 1.0f,  M, DMODEL, DMODEL);
    gemm_bf16_wmma<1><<<ggrid, 256, 0, stream>>>(Xb, Wvt, bv, (void*)Vt, 1.0f,  M, DMODEL, DMODEL);

    flash_attn_wmma<<<dim3(S_LEN / 128, BATCH * NHEAD), 256, 0, stream>>>(Qb, Kb, Vt, biasM, AO);

    gemm_bf16_wmma<2><<<ggrid, 256, 0, stream>>>(AO, Wot, bo, d_out, 1.0f, M, DMODEL, DMODEL);
}
